// ScaledDotProductAttention_72112500900228
// MI455X (gfx1250) — matrix operand order repaired and checked by reading, not compiled, not run on hardware
//
#include <hip/hip_runtime.h>

typedef __attribute__((ext_vector_type(16))) __bf16 v16bf;
typedef __attribute__((ext_vector_type(8)))  __bf16 v8bf;
typedef __attribute__((ext_vector_type(4)))  __bf16 v4bf;
typedef __attribute__((ext_vector_type(8)))  float  v8f;
typedef __attribute__((ext_vector_type(4)))  float  v4f;

namespace {
constexpr int kB = 4, kH = 16, kS = 2048, kD = 128;
constexpr int BLOCK_M = 128;          // q rows per workgroup: 8 waves x 16 rows
constexpr int BLOCK_N = 64;           // keys per iteration (4 key-tiles of 16)
constexpr int KSTR = kD + 8;          // 136 bf16 -> 272 B row stride (16B multiple)
constexpr int VSTR = BLOCK_N + 8;     // 72 bf16 -> 144 B row stride (16B multiple)
constexpr float kScale = 0.08838834764831845f;   // 1/sqrt(128)
constexpr float kLog2e = 1.4426950408889634f;
}

__global__ __launch_bounds__(256)
void fa_fwd_causal(const float* __restrict__ Qg, const float* __restrict__ Kg,
                   const float* __restrict__ Vg, float* __restrict__ Og)
{
  // K tile row-major [64 keys][128 d], V tile transposed [128 d][64 keys], bf16
  __shared__ __align__(16) __bf16 Ks[BLOCK_N * KSTR];   // 17408 B
  __shared__ __align__(16) __bf16 Vt[kD * VSTR];        // 18432 B

  const int tid  = threadIdx.x;
  const int wave = tid >> 5;      // 0..7 (wave32)
  const int lane = tid & 31;
  const int half = lane >> 4;     // which 16-lane half of the wave
  const int l16  = lane & 15;

  const int q_base = blockIdx.x * BLOCK_M;
  const size_t head = (size_t)blockIdx.y * kS * kD;
  const int wrow = q_base + wave * 16;        // first q row owned by this wave
  const int qrow = wrow + l16;                // this lane's q row (B/C column)

  // ---- Q fragments: B-matrix layout for S^T = K * Q^T, pre-scaled by 1/sqrt(D) ----
  // B(d, n=qrow): lane holds column n=l16; elements e -> d = a*32 + half*16 + e
  v16bf qf[4];
  {
    const float* qp = Qg + head + (size_t)qrow * kD;
    #pragma unroll
    for (int a = 0; a < 4; ++a) {
      const float* p = qp + a * 32 + half * 16;
      v4f x0 = *(const v4f*)(p + 0);
      v4f x1 = *(const v4f*)(p + 4);
      v4f x2 = *(const v4f*)(p + 8);
      v4f x3 = *(const v4f*)(p + 12);
      #pragma unroll
      for (int e = 0; e < 4; ++e) {
        qf[a][e + 0]  = (__bf16)(x0[e] * kScale);
        qf[a][e + 4]  = (__bf16)(x1[e] * kScale);
        qf[a][e + 8]  = (__bf16)(x2[e] * kScale);
        qf[a][e + 12] = (__bf16)(x3[e] * kScale);
      }
    }
  }

  // O^T accumulators: 8 tiles of 16(d) x 16(q); C-layout col = qrow = l16
  v8f o[8];
  #pragma unroll
  for (int dt = 0; dt < 8; ++dt)
    #pragma unroll
    for (int r = 0; r < 8; ++r) o[dt][r] = 0.0f;

  float m_i = -3.0e38f;   // running max for this lane's q row
  float l_i = 0.0f;       // running denom

  const int kv_blocks = (q_base + BLOCK_M) / BLOCK_N;   // causal upper bound
  for (int kb = 0; kb < kv_blocks; ++kb) {
    const int kv_base = kb * BLOCK_N;

    // ---- stage K (row-major) and V (transposed) tiles into LDS as bf16 ----
    {
      const float* kp = Kg + head + (size_t)kv_base * kD;
      #pragma unroll
      for (int it = 0; it < 8; ++it) {
        int u = tid + it * 256;           // 2048 float4 units over 64x128
        int row = u >> 5;
        int c4 = (u & 31) * 4;
        v4f x = *(const v4f*)(kp + row * kD + c4);
        v4bf y;
        y[0] = (__bf16)x[0]; y[1] = (__bf16)x[1];
        y[2] = (__bf16)x[2]; y[3] = (__bf16)x[3];
        *(v4bf*)(&Ks[row * KSTR + c4]) = y;   // ds_store_b64
      }
      const float* vp = Vg + head + (size_t)kv_base * kD;
      #pragma unroll
      for (int it = 0; it < 8; ++it) {
        int u = tid + it * 256;           // 2048 units: (d, 4-key group)
        int d = u & 127;
        int kg = (u >> 7) * 4;            // 0,4,...,60
        v4bf y;
        #pragma unroll
        for (int i = 0; i < 4; ++i) y[i] = (__bf16)vp[(size_t)(kg + i) * kD + d];
        *(v4bf*)(&Vt[d * VSTR + kg]) = y;     // transposed store, ds_store_b64
      }
    }
    if (kb + 1 < kv_blocks) {   // pull next K/V block toward L2 while we compute
      __builtin_prefetch(Kg + head + (size_t)(kv_base + BLOCK_N) * kD + tid * 32, 0, 1);
      __builtin_prefetch(Vg + head + (size_t)(kv_base + BLOCK_N) * kD + tid * 32, 0, 1);
    }
    __syncthreads();

    // Waves whose rows are entirely below this key block: nothing survives the
    // causal mask -> skip compute, keep barriers (wave-uniform branch).
    const bool active_blk = (kv_base <= wrow + 15);
    if (active_blk) {
      // ---- S^T = K * Q^T : four 16(key) x 16(q) tiles, fp32 accum ----
      // A(m=key, k=d): lane row = t*16 + l16; elements: d = a*32 + half*16 + e
      v8f st[4];
      #pragma unroll
      for (int t = 0; t < 4; ++t) {
        v8f acc;
        #pragma unroll
        for (int r = 0; r < 8; ++r) acc[r] = 0.0f;
        #pragma unroll
        for (int a = 0; a < 4; ++a) {
          const __bf16* ap = &Ks[(t * 16 + l16) * KSTR + a * 32 + half * 16];
          v8bf lo = *(const v8bf*)ap;            // ds_load_b128
          v8bf hi = *(const v8bf*)(ap + 8);
          v16bf af = __builtin_shufflevector(lo, hi,
                       0,1,2,3,4,5,6,7,8,9,10,11,12,13,14,15);
          acc = __builtin_amdgcn_wmma_f32_16x16x32_bf16(
                  false, af, false, qf[a], (short)0, acc, false, false);
        }
        st[t] = acc;
      }

      // ---- causal mask (wave-uniform guard; C row = key = t*16 + half*8 + r) ----
      if (kv_base + BLOCK_N - 1 > wrow) {
        #pragma unroll
        for (int t = 0; t < 4; ++t)
          #pragma unroll
          for (int r = 0; r < 8; ++r) {
            int key = kv_base + t * 16 + half * 8 + r;
            if (key > qrow) st[t][r] = -3.0e38f;
          }
      }

      // ---- online softmax: stats are per q column = per lane scalar ----
      float bmax = -3.0e38f;
      #pragma unroll
      for (int t = 0; t < 4; ++t)
        #pragma unroll
        for (int r = 0; r < 8; ++r) bmax = fmaxf(bmax, st[t][r]);
      bmax = fmaxf(bmax, __shfl_xor(bmax, 16, 32));
      const float mnew = fmaxf(m_i, bmax);
      const float corr = __builtin_amdgcn_exp2f((m_i - mnew) * kLog2e);
      float rsum = 0.0f;
      #pragma unroll
      for (int t = 0; t < 4; ++t)
        #pragma unroll
        for (int r = 0; r < 8; ++r) {
          float p = __builtin_amdgcn_exp2f((st[t][r] - mnew) * kLog2e);
          st[t][r] = p;
          rsum += p;
        }
      rsum += __shfl_xor(rsum, 16, 32);
      l_i = l_i * corr + rsum;
      m_i = mnew;
      #pragma unroll
      for (int dt = 0; dt < 8; ++dt)
        #pragma unroll
        for (int r = 0; r < 8; ++r) o[dt][r] *= corr;

      // ---- build P^T B-fragments (2 x 32 keys x 16 q) from C-layout via xor-16 ----
      // B element e needs local key = half*16 + e; C layout holds key = 8*half + r.
      v16bf pf[2];
      #pragma unroll
      for (int g = 0; g < 2; ++g) {
        #pragma unroll
        for (int r = 0; r < 8; ++r) {
          float p0 = st[2 * g + 0][r], p1 = st[2 * g + 1][r];
          float s0 = __shfl_xor(p0, 16, 32);
          float s1 = __shfl_xor(p1, 16, 32);
          pf[g][r]     = (__bf16)(half ? s1 : p0);   // local keys 0..7  / 16..23
          pf[g][r + 8] = (__bf16)(half ? p1 : s0);   // local keys 8..15 / 24..31
        }
      }

      // ---- O^T += V^T * P^T : 8 d-tiles x 2 key-chunks ----
      #pragma unroll
      for (int dt = 0; dt < 8; ++dt) {
        #pragma unroll
        for (int g = 0; g < 2; ++g) {
          const __bf16* ap = &Vt[(dt * 16 + l16) * VSTR + g * 32 + half * 16];
          v8bf lo = *(const v8bf*)ap;
          v8bf hi = *(const v8bf*)(ap + 8);
          v16bf af = __builtin_shufflevector(lo, hi,
                       0,1,2,3,4,5,6,7,8,9,10,11,12,13,14,15);
          o[dt] = __builtin_amdgcn_wmma_f32_16x16x32_bf16(
                    false, af, false, pf[g], (short)0, o[dt], false, false);
        }
      }
    }
    __syncthreads();
  }

  // ---- epilogue: O[q][d] = O^T / l ; lane owns row qrow, d = dt*16 + half*8 + r ----
  const float rinv = 1.0f / l_i;
  float* op = Og + head + (size_t)qrow * kD;
  #pragma unroll
  for (int dt = 0; dt < 8; ++dt) {
    v4f y0, y1;
    #pragma unroll
    for (int i = 0; i < 4; ++i) {
      y0[i] = o[dt][i] * rinv;
      y1[i] = o[dt][i + 4] * rinv;
    }
    *(v4f*)(op + dt * 16 + half * 8 + 0) = y0;   // global_store_b128
    *(v4f*)(op + dt * 16 + half * 8 + 4) = y1;
  }
}

extern "C" void kernel_launch(void* const* d_in, const int* in_sizes, int n_in,
                              void* d_out, int out_size, void* d_ws, size_t ws_size,
                              hipStream_t stream) {
  (void)in_sizes; (void)n_in; (void)out_size; (void)d_ws; (void)ws_size;
  const float* Q = (const float*)d_in[0];
  const float* K = (const float*)d_in[1];
  const float* V = (const float*)d_in[2];
  // d_in[3] is the causal mask; the causal structure is hardcoded in-kernel.
  float* O = (float*)d_out;
  dim3 grid(kS / BLOCK_M, kB * kH);   // 16 q-tiles x 64 (b,h) heads = 1024 WGs
  fa_fwd_causal<<<grid, 256, 0, stream>>>(Q, K, V, O);
}
